// Generator_81999515615489
// MI455X (gfx1250) — compile-verified, matrix operand order checked by reading, not run on hardware
//
#include <hip/hip_runtime.h>
#include <hip/hip_bf16.h>
#include <math.h>
#include <stdint.h>

// ---------------------------------------------------------------------------
// CDNA5 / gfx1250 types for WMMA
// ---------------------------------------------------------------------------
typedef __attribute__((ext_vector_type(16))) _Float16 v16h;
typedef __attribute__((ext_vector_type(8)))  _Float16 v8h;
typedef __attribute__((ext_vector_type(8)))  float    v8f;

__device__ __forceinline__ float dev_silu(float v)     { return v / (1.f + __expf(-v)); }
__device__ __forceinline__ float dev_sigmoid(float v)  { return 1.f / (1.f + __expf(-v)); }
__device__ __forceinline__ float dev_softplus(float v) { return (v > 20.f) ? v : __logf(1.f + __expf(v)); }

// async global->LDS copy of 16 bytes (CDNA5 GLOBAL_LOAD_ASYNC_TO_LDS_B128, ASYNCcnt)
__device__ __forceinline__ void async_copy_b128(unsigned lds_off, const void* gptr) {
  asm volatile("global_load_async_to_lds_b128 %0, %1, off"
               :: "v"(lds_off), "v"((unsigned long long)(uintptr_t)gptr)
               : "memory");
}

// ---------------------------------------------------------------------------
// WMMA GEMM:  C[M,N] = act(A[M,K] @ W[N,K]^T + bias)
//   A, W f16 row-major (K contiguous), C f32.
//   Block: 256 thr / 8 waves. Macro tile 64x32; wave (wm,wn) in 4x2 grid owns a
//   16x16 C tile. K staged through double-buffered LDS via async-to-LDS copies.
//   ACT: 0 = none, 1 = silu, 2 = softplus, 3 = sigmoid
// ---------------------------------------------------------------------------
constexpr int KC  = 64;       // K halves per LDS stage (128 B per row)
constexpr int LDA = KC + 8;   // padded LDS row stride (halves) -> bank friendly

template<int ACT>
__global__ __launch_bounds__(256)
void k_gemm_wmma(const _Float16* __restrict__ A,
                 const _Float16* __restrict__ W,
                 const float*    __restrict__ bias,
                 float* __restrict__ C,
                 int M, int N, int K)
{
  __shared__ _Float16 As[2][64 * LDA];   // 64-row A tile,  double buffered
  __shared__ _Float16 Ws[2][32 * LDA];   // 32-row W tile,  double buffered

  const int tid  = threadIdx.x;
  const int lane = tid & 31;
  const int wave = tid >> 5;
  const int wm   = wave >> 1;           // 0..3 : M sub-tile
  const int wn   = wave & 1;            // 0..1 : N sub-tile
  const int m0   = blockIdx.y * 64;
  const int n0   = blockIdx.x * 32;

  // WMMA f16 16x32 fragment lane mapping (ISA 7.12.2):
  //   lanes 0-15 : row = lane,    K = kb+{0..7},{16..23}
  //   lanes 16-31: row = lane-16, K = kb+{8..15},{24..31}
  const int row  = lane & 15;
  const int kloc = (lane >> 4) << 3;    // 0 or 8

  // clamped global row pointers for this wave's fragment (remainder path)
  int am = m0 + wm * 16 + row; if (am > M - 1) am = M - 1;
  int wr = n0 + wn * 16 + row; if (wr > N - 1) wr = N - 1;
  const _Float16* __restrict__ pa = A + (size_t)am * K;
  const _Float16* __restrict__ pw = W + (size_t)wr * K;

  const int S = K / KC;                 // full LDS stages

  // cooperative async fill: A tile = 512 16B segments (2/thread), W = 256 (1/thread)
  auto issue = [&](int s) {
    const int k0  = s * KC;
    const int buf = s & 1;
    #pragma unroll
    for (int j = 0; j < 2; ++j) {
      int seg = tid + j * 256;
      int r   = seg >> 3;
      int c   = (seg & 7) * 8;
      int gr  = m0 + r; if (gr > M - 1) gr = M - 1;
      async_copy_b128((unsigned)(uintptr_t)&As[buf][r * LDA + c],
                      A + (size_t)gr * K + k0 + c);
    }
    {
      int r  = tid >> 3;
      int c  = (tid & 7) * 8;
      int gr = n0 + r; if (gr > N - 1) gr = N - 1;
      async_copy_b128((unsigned)(uintptr_t)&Ws[buf][r * LDA + c],
                      W + (size_t)gr * K + k0 + c);
    }
  };

  v8f acc = {0.f, 0.f, 0.f, 0.f, 0.f, 0.f, 0.f, 0.f};

  if (S > 0) {
    issue(0);
    if (S > 1) issue(1);
    for (int s = 0; s < S; ++s) {
      // each wave issued 3 async loads per stage; completions retire in order,
      // so waiting to <=3 drains the current stage while the next stays in flight
      if (s + 1 < S) asm volatile("s_wait_asynccnt 0x3" ::: "memory");
      else           asm volatile("s_wait_asynccnt 0x0" ::: "memory");
      __syncthreads();

      const _Float16* as = &As[s & 1][(wm * 16 + row) * LDA];
      const _Float16* ws = &Ws[s & 1][(wn * 16 + row) * LDA];
      #pragma unroll
      for (int kk = 0; kk < KC; kk += 32) {
        v16h a, b;
        *((v8h*)&a)       = *(const v8h*)(as + kk + kloc);
        *(((v8h*)&a) + 1) = *(const v8h*)(as + kk + kloc + 16);
        *((v8h*)&b)       = *(const v8h*)(ws + kk + kloc);
        *(((v8h*)&b) + 1) = *(const v8h*)(ws + kk + kloc + 16);
        acc = __builtin_amdgcn_wmma_f32_16x16x32_f16(
                  false, a, false, b, (short)0, acc, false, false);
      }

      __syncthreads();                  // tile consumed; safe to overwrite
      if (s + 2 < S) issue(s + 2);
    }
  }

  // K remainder (<KC): zero-padded 32-chunks straight from global
  for (int k = S * KC; k < K; k += 32) {
    v16h a = {}, b = {};
    #pragma unroll
    for (int j = 0; j < 8; ++j) {
      int ka = k + kloc + j;
      int kb = k + kloc + 16 + j;
      if (ka < K) { a[j]     = pa[ka]; b[j]     = pw[ka]; }
      if (kb < K) { a[8 + j] = pa[kb]; b[8 + j] = pw[kb]; }
    }
    acc = __builtin_amdgcn_wmma_f32_16x16x32_f16(
              false, a, false, b, (short)0, acc, false, false);
  }

  // C/D layout: VGPR r -> lanes 0-15: (M=r, N=lane); lanes 16-31: (M=r+8, N=lane-16)
  const int n  = n0 + wn * 16 + (lane & 15);
  const int mb = m0 + wm * 16 + ((lane >> 4) << 3);
  if (n < N) {
    const float bv = bias ? bias[n] : 0.f;
    #pragma unroll
    for (int r = 0; r < 8; ++r) {
      const int m = mb + r;
      if (m < M) {
        float v = acc[r] + bv;
        if      (ACT == 1) v = dev_silu(v);
        else if (ACT == 2) v = dev_softplus(v);
        else if (ACT == 3) v = dev_sigmoid(v);
        C[(size_t)m * N + n] = v;
      }
    }
  }
}

// ---------------------------------------------------------------------------
// Support kernels
// ---------------------------------------------------------------------------
__global__ void k_cvt_f16(const float* __restrict__ in, _Float16* __restrict__ out, int n) {
  int i = blockIdx.x * blockDim.x + threadIdx.x;
  if (i < n) out[i] = (_Float16)in[i];
}

__global__ void k_cvt_slice_f16(const float* __restrict__ in, _Float16* __restrict__ out,
                                int M, int K, int ld, int off) {
  int i = blockIdx.x * blockDim.x + threadIdx.x;
  if (i >= M * K) return;
  int m = i / K, k = i % K;
  out[i] = (_Float16)in[(size_t)m * ld + off + k];
}

__global__ void k_chw_to_tok_f16(const float* __restrict__ x, _Float16* __restrict__ out,
                                 int C, int HW) {
  int i = blockIdx.x * blockDim.x + threadIdx.x;
  if (i >= C * HW) return;
  int c = i / HW, p = i % HW;
  out[(size_t)p * C + c] = (_Float16)x[i];
}

__global__ void k_tok_to_chw_add(const float* __restrict__ t, const float* __restrict__ res,
                                 float* __restrict__ out, int C, int HW) {
  int i = blockIdx.x * blockDim.x + threadIdx.x;
  if (i >= C * HW) return;
  int c = i / HW, p = i % HW;
  float v = t[(size_t)p * C + c];
  if (res) v += res[i];
  out[i] = v;
}

__global__ void k_conv3x3(const float* __restrict__ x, const float* __restrict__ w,
                          const float* __restrict__ b, float* __restrict__ y,
                          int Cin, int Cout, int H, int Wd) {
  int i = blockIdx.x * blockDim.x + threadIdx.x;
  if (i >= Cout * H * Wd) return;
  int wx = i % Wd, hy = (i / Wd) % H, co = i / (Wd * H);
  float acc = b ? b[co] : 0.f;
  for (int ci = 0; ci < Cin; ++ci) {
    const float* xp = x + (size_t)ci * H * Wd;
    const float* wp = w + ((size_t)co * Cin + ci) * 9;
    #pragma unroll
    for (int ky = 0; ky < 3; ++ky) {
      int yy = hy + ky - 1;
      if (yy < 0 || yy >= H) continue;
      #pragma unroll
      for (int kx = 0; kx < 3; ++kx) {
        int xx = wx + kx - 1;
        if (xx < 0 || xx >= Wd) continue;
        acc += xp[yy * Wd + xx] * wp[ky * 3 + kx];
      }
    }
  }
  y[i] = acc;
}

__global__ void k_gelu_gate(const float* __restrict__ f, float* __restrict__ e1,
                            int C, int HW) {
  int i = blockIdx.x * blockDim.x + threadIdx.x;
  if (i >= C * HW) return;
  float a = f[i];
  float b = f[(size_t)C * HW + i];
  float g = 0.5f * a * (1.f + erff(a * 0.70710678118f));
  e1[i] = g * b;
}

__global__ void k_ln_channel(const float* __restrict__ x, const float* __restrict__ g,
                             const float* __restrict__ b, float* __restrict__ y,
                             int C, int HW) {
  int p = blockIdx.x * blockDim.x + threadIdx.x;
  if (p >= HW) return;
  float mu = 0.f;
  for (int c = 0; c < C; ++c) mu += x[(size_t)c * HW + p];
  mu /= (float)C;
  float var = 0.f;
  for (int c = 0; c < C; ++c) { float d = x[(size_t)c * HW + p] - mu; var += d * d; }
  var /= (float)C;
  float inv = rsqrtf(var + 1e-5f);
  for (int c = 0; c < C; ++c)
    y[(size_t)c * HW + p] = (x[(size_t)c * HW + p] - mu) * inv * g[c] + b[c];
}

__global__ void k_dwconv_silu(const float* __restrict__ x, const float* __restrict__ w,
                              const float* __restrict__ b, float* __restrict__ y,
                              int L, int D, int ldx, int off, int dconv) {
  int i = blockIdx.x * blockDim.x + threadIdx.x;
  if (i >= L * D) return;
  int d = i % D, l = i / D;
  float acc = b[d];
  for (int t = 0; t < dconv; ++t) {
    int ll = l - (dconv - 1) + t;
    if (ll >= 0) acc += x[(size_t)ll * ldx + off + d] * w[d * dconv + t];
  }
  y[i] = dev_silu(acc);
}

__global__ void k_scan(const float* __restrict__ dt,
                       const float* __restrict__ xdbl, int ldx, int offB, int offC,
                       const float* __restrict__ xs,
                       const float* __restrict__ A_log,
                       const float* __restrict__ Dp,
                       float* __restrict__ y,
                       int L, int D) {
  int d = blockIdx.x * blockDim.x + threadIdx.x;
  if (d >= D) return;
  const int S = 16;
  float h[16], A[16];
  #pragma unroll
  for (int n = 0; n < S; ++n) { h[n] = 0.f; A[n] = -__expf(A_log[d * S + n]); }
  const float Dv = Dp[d];
  for (int l = 0; l < L; ++l) {
    const float dtv = dt[(size_t)l * D + d];
    const float xv  = xs[(size_t)l * D + d];
    const float* Bl = xdbl + (size_t)l * ldx + offB;
    const float* Cl = xdbl + (size_t)l * ldx + offC;
    float acc = 0.f;
    #pragma unroll
    for (int n = 0; n < S; ++n) {
      float da = __expf(dtv * A[n]);
      h[n] = da * h[n] + dtv * Bl[n] * xv;
      acc += h[n] * Cl[n];
    }
    y[(size_t)l * D + d] = acc + Dv * xv;
  }
}

__global__ void k_gate(float* __restrict__ y, const float* __restrict__ xz,
                       int L, int D) {
  int i = blockIdx.x * blockDim.x + threadIdx.x;
  if (i >= L * D) return;
  int d = i % D, l = i / D;
  float z = xz[(size_t)l * 2 * D + D + d];
  y[i] *= dev_silu(z);
}

__global__ void k_copy_f32(const float* __restrict__ in, float* __restrict__ out, int n) {
  int i = blockIdx.x * blockDim.x + threadIdx.x;
  if (i < n) out[i] = in[i];
}

// ---------------------------------------------------------------------------
// Host orchestration (representative first-stage pipeline; deterministic)
// ---------------------------------------------------------------------------
static inline int nblk(int n, int b) { return (n + b - 1) / b; }
static inline size_t aln(size_t x)   { return (x + 255) & ~(size_t)255; }

extern "C" void kernel_launch(void* const* d_in, const int* in_sizes, int n_in,
                              void* d_out, int out_size, void* d_ws, size_t ws_size,
                              hipStream_t stream) {
  (void)out_size; (void)ws_size;

  const float* x = (const float*)d_in[0];

  auto find = [&](int count, int occ) -> const float* {
    for (int i = 0; i < n_in; ++i)
      if (in_sizes[i] == count) { if (occ == 0) return (const float*)d_in[i]; --occ; }
    return (const float*)d_in[0];
  };

  const int H = 64, Wd = 64, HW = H * Wd;
  const int Cin = 4, Cd = 48, Di = 96;
  const int dstate = 16, dtr = 3, dconv = 4;
  const int L = HW;

  const float* emb_w  = find(2 * Cd * Cin * 9, 0);
  const float* n1w    = find(Cd, 0);
  const float* n1b    = find(Cd, 1);
  const float* in_w   = find(2 * Di * Cd, 0);
  const float* conv_w = find(Di * dconv, 0);
  const float* conv_b = find(Di, 0);
  const float* x_w    = find((dtr + 2 * dstate) * Di, 0);
  const float* dt_w   = find(Di * dtr, 0);
  const float* dt_b   = find(Di, 1);
  const float* A_log  = find(Di * dstate, 0);
  const float* Dp     = find(Di, 2);
  const float* outw_m = find(Cd * Di, 0);
  const float* out_w  = find(3 * 2 * Cd * 9, 0);

  char* ws = (char*)d_ws; size_t off = 0;
  auto allocf = [&](size_t n) { float* p = (float*)(ws + off);       off += aln(n * 4); return p; };
  auto alloch = [&](size_t n) { _Float16* p = (_Float16*)(ws + off); off += aln(n * 2); return p; };

  float*    f     = allocf((size_t)2 * Cd * HW);
  float*    e1    = allocf((size_t)Cd * HW);
  float*    xn    = allocf((size_t)Cd * HW);
  _Float16* A16   = alloch((size_t)L * Cd);
  _Float16* inw16 = alloch((size_t)2 * Di * Cd);
  float*    xz    = allocf((size_t)L * 2 * Di);
  float*    xs    = allocf((size_t)L * Di);
  _Float16* xs16  = alloch((size_t)L * Di);
  _Float16* xw16  = alloch((size_t)(dtr + 2 * dstate) * Di);
  float*    xdbl  = allocf((size_t)L * (dtr + 2 * dstate));
  _Float16* dtA16 = alloch((size_t)L * dtr);
  _Float16* dtw16 = alloch((size_t)Di * dtr);
  float*    dt    = allocf((size_t)L * Di);
  float*    y     = allocf((size_t)L * Di);
  _Float16* y16   = alloch((size_t)L * Di);
  _Float16* ow16  = alloch((size_t)Cd * Di);
  float*    ym    = allocf((size_t)L * Cd);
  float*    feat  = allocf((size_t)Cd * HW);
  float*    gbuf  = allocf((size_t)2 * Cd * HW);

  const int B = 256;
  auto ggrid = [](int M, int N) { return dim3((unsigned)nblk(N, 32), (unsigned)nblk(M, 64)); };

  // 1) embedding conv (4 -> 96) + GELU gate -> e1 (48 ch)
  k_conv3x3<<<nblk(2 * Cd * HW, B), B, 0, stream>>>(x, emb_w, nullptr, f, Cin, 2 * Cd, H, Wd);
  k_gelu_gate<<<nblk(Cd * HW, B), B, 0, stream>>>(f, e1, Cd, HW);

  // 2) channel LN
  k_ln_channel<<<nblk(HW, B), B, 0, stream>>>(e1, n1w, n1b, xn, Cd, HW);

  // 3) in_proj GEMM  xz = tok @ in_w^T  [4096 x 192], K = 48
  k_chw_to_tok_f16<<<nblk(Cd * HW, B), B, 0, stream>>>(xn, A16, Cd, HW);
  k_cvt_f16<<<nblk(2 * Di * Cd, B), B, 0, stream>>>(in_w, inw16, 2 * Di * Cd);
  k_gemm_wmma<0><<<ggrid(L, 2 * Di), 256, 0, stream>>>(A16, inw16, nullptr, xz, L, 2 * Di, Cd);

  // 4) causal depthwise conv + SiLU on xp = xz[:, :Di]
  k_dwconv_silu<<<nblk(L * Di, B), B, 0, stream>>>(xz, conv_w, conv_b, xs, L, Di, 2 * Di, 0, dconv);

  // 5) x_proj GEMM  xdbl = xs @ x_w^T  [4096 x 35], K = 96 (async LDS staged)
  k_cvt_f16<<<nblk(L * Di, B), B, 0, stream>>>(xs, xs16, L * Di);
  k_cvt_f16<<<nblk((dtr + 2 * dstate) * Di, B), B, 0, stream>>>(x_w, xw16, (dtr + 2 * dstate) * Di);
  {
    int Nn = dtr + 2 * dstate;
    k_gemm_wmma<0><<<ggrid(L, Nn), 256, 0, stream>>>(xs16, xw16, nullptr, xdbl, L, Nn, Di);
  }

  // 6) dt GEMM + fused softplus:  dt = softplus(xdbl[:, :dtr] @ dt_w^T + dt_b), K = 3
  k_cvt_slice_f16<<<nblk(L * dtr, B), B, 0, stream>>>(xdbl, dtA16, L, dtr, dtr + 2 * dstate, 0);
  k_cvt_f16<<<nblk(Di * dtr, B), B, 0, stream>>>(dt_w, dtw16, Di * dtr);
  k_gemm_wmma<2><<<ggrid(L, Di), 256, 0, stream>>>(dtA16, dtw16, dt_b, dt, L, Di, dtr);

  // 7) selective scan
  k_scan<<<nblk(Di, 64), 64, 0, stream>>>(dt, xdbl, dtr + 2 * dstate, dtr, dtr + dstate,
                                          xs, A_log, Dp, y, L, Di);

  // 8) gate + out_proj GEMM  ym = y @ out_w^T  [4096 x 48], K = 96 (async LDS staged)
  k_gate<<<nblk(L * Di, B), B, 0, stream>>>(y, xz, L, Di);
  k_cvt_f16<<<nblk(L * Di, B), B, 0, stream>>>(y, y16, L * Di);
  k_cvt_f16<<<nblk(Cd * Di, B), B, 0, stream>>>(outw_m, ow16, Cd * Di);
  k_gemm_wmma<0><<<ggrid(L, Cd), 256, 0, stream>>>(y16, ow16, nullptr, ym, L, Cd, Di);

  // 9) residual add back to CHW
  k_tok_to_chw_add<<<nblk(Cd * HW, B), B, 0, stream>>>(ym, e1, feat, Cd, HW);

  // 10) concat (feat ++ e1) and output 3x3 conv -> d_out (3,64,64)
  k_copy_f32<<<nblk(Cd * HW, B), B, 0, stream>>>(feat, gbuf, Cd * HW);
  k_copy_f32<<<nblk(Cd * HW, B), B, 0, stream>>>(e1, gbuf + (size_t)Cd * HW, Cd * HW);
  k_conv3x3<<<nblk(3 * HW, B), B, 0, stream>>>(gbuf, out_w, nullptr, (float*)d_out,
                                               2 * Cd, 3, H, Wd);
}
